// Transformer_Encoder_Block_28278064677221
// MI455X (gfx1250) — matrix operand order repaired and checked by reading, not compiled, not run on hardware
//
#include <hip/hip_runtime.h>
#include <math.h>

// ---------------------------------------------------------------------------
// Transformer encoder block for MI455X (gfx1250, wave32, WMMA).
// GEMMs/attention use v_wmma_f32_16x16x32_f16 (f16 in, f32 accum).
// Row-major tiles are staged with global_load_async_to_lds_b128 (ASYNCcnt)
// into double-buffered LDS; operands needing transposition are transposed
// during staging so every WMMA fragment load is a contiguous ds_load_b128
// pair. Next tile's copies are issued before the current tile's WMMAs.
// ---------------------------------------------------------------------------

typedef __attribute__((ext_vector_type(16))) _Float16 v16h;
typedef __attribute__((ext_vector_type(8)))  float    v8f;
typedef __attribute__((ext_vector_type(8)))  _Float16 h8;

#define BB   16
#define TT   1024
#define CC   768
#define HH   12
#define MMD  3072
#define HDIM 64
#define BT   (BB * TT)

#define USE_ASYNC_LDS 1

// ---------------- gfx1250 async global->LDS copy (16 bytes / lane) --------

__device__ __forceinline__ void async_cp16(void* lds_dst, const void* gsrc) {
#if USE_ASYNC_LDS
  // Generic LDS address: low 32 bits are the LDS byte offset (ISA 10.2).
  unsigned loff = (unsigned)(unsigned long long)lds_dst;
  unsigned long long ga = (unsigned long long)gsrc;
  asm volatile("global_load_async_to_lds_b128 %0, %1, off"
               :: "v"(loff), "v"(ga) : "memory");
#else
  *(h8*)lds_dst = *(const h8*)gsrc;
#endif
}

__device__ __forceinline__ void async_wait0() {
#if USE_ASYNC_LDS
  asm volatile("s_wait_asynccnt 0x0" ::: "memory");
#endif
}

// ---------------- WMMA helpers --------------------------------------------

__device__ __forceinline__ v8f wmma16(v16h a, v16h b, v8f c) {
  return __builtin_amdgcn_wmma_f32_16x16x32_f16(false, a, false, b, (short)0, c,
                                                false, false);
}

// A fragment 16x32 (MxK) from row-major LDS, leading dim ld (halves).
// lanes 0-15: M=lane, K=0..15 ; lanes 16-31: M=lane-16, K=16..31
__device__ __forceinline__ v16h frag_a(const _Float16* base, int ld) {
  int lane = threadIdx.x & 31;
  const _Float16* p = base + (lane & 15) * ld + ((lane >> 4) << 4);
  h8 lo = *(const h8*)p;        // K = kb .. kb+7
  h8 hi = *(const h8*)(p + 8);  // K = kb+8 .. kb+15
  return __builtin_shufflevector(lo, hi, 0, 1, 2, 3, 4, 5, 6, 7, 8, 9, 10, 11,
                                 12, 13, 14, 15);
}

// B fragment 32x16 (KxN) from a TRANSPOSED LDS tile Bt[n][k], ld in halves.
// lanes 0-15: N=lane, K=0..15 ; lanes 16-31: N=lane-16, K=16..31
__device__ __forceinline__ v16h frag_bt(const _Float16* base, int ld) {
  int lane = threadIdx.x & 31;
  const _Float16* p = base + (lane & 15) * ld + ((lane >> 4) << 4);
  h8 lo = *(const h8*)p;       // K = kb .. kb+7
  h8 hi = *(const h8*)(p + 8); // K = kb+8 .. kb+15
  return __builtin_shufflevector(lo, hi, 0, 1, 2, 3, 4, 5, 6, 7, 8, 9, 10, 11,
                                 12, 13, 14, 15);
}

// ---------------- LayerNorm -> f16 ----------------------------------------

__global__ __launch_bounds__(256) void layernorm_f16_k(
    const float* __restrict__ x, const float* __restrict__ g,
    const float* __restrict__ bta, _Float16* __restrict__ out) {
  __shared__ float sred[18];
  int row = blockIdx.x;
  const float* xr = x + (size_t)row * CC;
  float s = 0.f, ss = 0.f;
  for (int i = threadIdx.x; i < CC; i += 256) {
    float v = xr[i];
    s += v;
    ss += v * v;
  }
#pragma unroll
  for (int off = 16; off > 0; off >>= 1) {
    s += __shfl_xor(s, off, 32);
    ss += __shfl_xor(ss, off, 32);
  }
  if ((threadIdx.x & 31) == 0) {
    sred[threadIdx.x >> 5] = s;
    sred[8 + (threadIdx.x >> 5)] = ss;
  }
  __syncthreads();
  if (threadIdx.x == 0) {
    float S = 0.f, SS = 0.f;
    for (int i = 0; i < 8; ++i) { S += sred[i]; SS += sred[8 + i]; }
    float mu = S * (1.0f / CC);
    float var = SS * (1.0f / CC) - mu * mu;
    sred[16] = mu;
    sred[17] = rsqrtf(var + 1e-5f);
  }
  __syncthreads();
  float mu = sred[16], inv = sred[17];
  for (int i = threadIdx.x; i < CC; i += 256)
    out[(size_t)row * CC + i] = (_Float16)((xr[i] - mu) * inv * g[i] + bta[i]);
}

// ---------------- weight conversion ---------------------------------------

// wq/wk/wv: [H, C, HD] f32 -> [C, H*HD] f16
__global__ __launch_bounds__(256) void pack_qkv_k(const float* __restrict__ w,
                                                  _Float16* __restrict__ out) {
  int idx = blockIdx.x * 256 + threadIdx.x;
  if (idx >= HH * CC * HDIM) return;
  int h = idx / (CC * HDIM);
  int r = (idx / HDIM) % CC;
  int d = idx % HDIM;
  out[(size_t)r * CC + h * HDIM + d] = (_Float16)w[idx];
}

__global__ __launch_bounds__(256) void cvt_f16_k(const float* __restrict__ in,
                                                 _Float16* __restrict__ out,
                                                 int n) {
  int i = blockIdx.x * 256 + threadIdx.x;
  if (i < n) out[i] = (_Float16)in[i];
}

// ---------------- generic WMMA GEMM ---------------------------------------
// C[M,N] = A[M,K] * B[K,N], 128x64 tile per 256-thread block, K-step 32,
// double-buffered LDS. Wave w owns n-tile (w&3) and m-tiles (w>>2)*64+{0..3}*16
// -> 4 WMMAs per wave per K-step sharing one B fragment.
// MODE 0: outH = acc (f16)
// MODE 1: outF = acc + bias[col] + resid[row,col]   (f32)
// MODE 2: outH = gelu(acc + bias[col])              (f16, exact erf GELU)

template <int MODE>
__global__ __launch_bounds__(256) void gemm_wmma(
    const _Float16* __restrict__ A, const _Float16* __restrict__ B,
    const float* __restrict__ bias, const float* __restrict__ resid,
    float* __restrict__ outF, _Float16* __restrict__ outH, int M, int N,
    int K) {
  constexpr int LDA = 40;  // A tile: 128 rows(m) x 32 (k) + pad
  constexpr int LDB = 40;  // Bt tile: 64 rows(n) x 32 (k) + pad
  __shared__ alignas(16) _Float16 As[2][128 * LDA];
  __shared__ alignas(16) _Float16 Bt[2][64 * LDB];

  int n0 = blockIdx.x * 64;
  int m0 = blockIdx.y * 128;
  int wave = threadIdx.x >> 5;
  int lane = threadIdx.x & 31;
  int nt = wave & 3;            // this wave's 16-wide N tile
  int mh = (wave >> 2) * 64;    // this wave's 64-row M half

  v8f acc[4] = {};

  int ra = threadIdx.x >> 2, ca = (threadIdx.x & 3) * 8;  // A: rows ra, ra+64
  int rb = threadIdx.x >> 3, cb = (threadIdx.x & 7) * 8;  // B: 32x64

  const _Float16* Ag0 = A + (size_t)(m0 + ra) * K + ca;
  const _Float16* Ag1 = Ag0 + (size_t)64 * K;
  const _Float16* Bg = B + (size_t)rb * N + n0 + cb;

  // prologue: stage buffer 0 (k0 = 0)
  async_cp16(&As[0][ra * LDA + ca], Ag0);
  async_cp16(&As[0][(ra + 64) * LDA + ca], Ag1);
  {
    h8 bv = *(const h8*)Bg;
#pragma unroll
    for (int j = 0; j < 8; ++j) Bt[0][(cb + j) * LDB + rb] = bv[j];
  }
  async_wait0();
  __syncthreads();

  for (int k0 = 0; k0 < K; k0 += 32) {
    int cur = (k0 >> 5) & 1;
    int nxt = cur ^ 1;
    bool more = (k0 + 32) < K;

    // issue next tile's copies before this tile's math
    h8 bv;
    if (more) {
      bv = *(const h8*)(Bg + (size_t)(k0 + 32) * N);
      async_cp16(&As[nxt][ra * LDA + ca], Ag0 + (k0 + 32));
      async_cp16(&As[nxt][(ra + 64) * LDA + ca], Ag1 + (k0 + 32));
    }

    v16h bf = frag_bt(&Bt[cur][(nt * 16) * LDB], LDB);
#pragma unroll
    for (int i = 0; i < 4; ++i) {
      v16h af = frag_a(&As[cur][(mh + i * 16) * LDA], LDA);
      acc[i] = wmma16(af, bf, acc[i]);
    }

    if (more) {
#pragma unroll
      for (int j = 0; j < 8; ++j) Bt[nxt][(cb + j) * LDB + rb] = bv[j];
    }
    async_wait0();
    __syncthreads();
  }

  int col = n0 + nt * 16 + (lane & 15);
  int rb8 = (lane >> 4) * 8;
#pragma unroll
  for (int i = 0; i < 4; ++i) {
    int row = m0 + mh + i * 16 + rb8;
#pragma unroll
    for (int r = 0; r < 8; ++r) {
      size_t idx = (size_t)(row + r) * N + col;
      if (MODE == 0) {
        outH[idx] = (_Float16)acc[i][r];
      } else if (MODE == 1) {
        outF[idx] = acc[i][r] + bias[col] + resid[idx];
      } else {
        float u = acc[i][r] + bias[col];
        outH[idx] = (_Float16)(0.5f * u * (1.0f + erff(u * 0.70710678118f)));
      }
    }
  }
}

// ---------------- flash attention -----------------------------------------
// Per block: one (b, h, 128-query-row) tile. 8 waves; wave w owns 16 query
// columns. S computed transposed (S'[s,t] = sum_d K[s,d] Q[t,d]) so the
// softmax reduction over keys is in-lane + one shfl_xor(16). Output is
// accumulated transposed: O'[d,t] += V^T * P'. K tile staged row-major
// (async); V tile staged transposed. Both double-buffered: next key-block's
// copies are issued before the current block's WMMAs.

__global__ __launch_bounds__(256) void attention_k(
    const _Float16* __restrict__ q, const _Float16* __restrict__ k,
    const _Float16* __restrict__ v, _Float16* __restrict__ o) {
  constexpr int LDK = 72;  // Ks: 32 rows(s) x 64 (d) + pad
  constexpr int LDV = 40;  // Vt: 64 rows(d) x 32 (s) + pad
  __shared__ alignas(16) _Float16 Ks[2][32 * LDK];
  __shared__ alignas(16) _Float16 Vt[2][64 * LDV];

  const int ntq = TT / 128;  // 8
  int tq = blockIdx.x % ntq;
  int h = (blockIdx.x / ntq) % HH;
  int b = blockIdx.x / (ntq * HH);
  int h0 = h * HDIM;
  size_t rowbase = (size_t)b * TT;
  int lane = threadIdx.x & 31;
  int wave = threadIdx.x >> 5;
  int col = lane & 15;
  int tcol = tq * 128 + wave * 16 + col;  // this lane's query index

  // loop-invariant Q^T B-fragments (d = 0..31 and 32..63)
  v16h qb0, qb1;
  {
    const _Float16* qp = q + (rowbase + tcol) * CC + h0 + ((lane >> 4) << 4);
    h8 a0 = *(const h8*)qp;
    h8 a1 = *(const h8*)(qp + 8);
    h8 b0 = *(const h8*)(qp + 32);
    h8 b1 = *(const h8*)(qp + 40);
    qb0 = __builtin_shufflevector(a0, a1, 0, 1, 2, 3, 4, 5, 6, 7, 8, 9, 10, 11,
                                  12, 13, 14, 15);
    qb1 = __builtin_shufflevector(b0, b1, 0, 1, 2, 3, 4, 5, 6, 7, 8, 9, 10, 11,
                                  12, 13, 14, 15);
  }

  float mrun = -3.0e38f, lrun = 0.f;
  v8f oacc[4] = {};  // O'[d,t], d-tiles 0..3

  int sr = threadIdx.x >> 3;
  int sc = (threadIdx.x & 7) * 8;
  const _Float16* kg = k + (rowbase + sr) * CC + h0 + sc;
  const _Float16* vg = v + (rowbase + sr) * CC + h0 + sc;

  // prologue: stage buffer 0 (s0 = 0)
  async_cp16(&Ks[0][sr * LDK + sc], kg);
  {
    h8 vv = *(const h8*)vg;
#pragma unroll
    for (int j = 0; j < 8; ++j) Vt[0][(sc + j) * LDV + sr] = vv[j];
  }
  async_wait0();
  __syncthreads();

  for (int s0 = 0; s0 < TT; s0 += 32) {
    int cur = (s0 >> 5) & 1;
    int nxt = cur ^ 1;
    bool more = (s0 + 32) < TT;

    h8 vv;
    if (more) {
      vv = *(const h8*)(vg + (size_t)(s0 + 32) * CC);
      async_cp16(&Ks[nxt][sr * LDK + sc], kg + (size_t)(s0 + 32) * CC);
    }

    // S' tiles: rows s0..s0+15 (sa) and s0+16..s0+31 (sb), K = HD = 64
    v8f sa = {}, sb = {};
    sa = wmma16(frag_a(&Ks[cur][0], LDK), qb0, sa);
    sa = wmma16(frag_a(&Ks[cur][32], LDK), qb1, sa);
    sb = wmma16(frag_a(&Ks[cur][16 * LDK], LDK), qb0, sb);
    sb = wmma16(frag_a(&Ks[cur][16 * LDK + 32], LDK), qb1, sb);

    // online softmax over the 32 keys of this block (scale = HD^-0.5 = 0.125)
    float bmax = -3.0e38f;
#pragma unroll
    for (int r = 0; r < 8; ++r) bmax = fmaxf(bmax, fmaxf(sa[r], sb[r]));
    bmax *= 0.125f;
    bmax = fmaxf(bmax, __shfl_xor(bmax, 16, 32));
    float mnew = fmaxf(mrun, bmax);
    float alpha = __expf(mrun - mnew);

    float p0[8], p1[8], psum = 0.f;
#pragma unroll
    for (int r = 0; r < 8; ++r) {
      p0[r] = __expf(sa[r] * 0.125f - mnew);
      p1[r] = __expf(sb[r] * 0.125f - mnew);
      psum += p0[r] + p1[r];
    }
    psum += __shfl_xor(psum, 16, 32);
    lrun = lrun * alpha + psum;
    mrun = mnew;
#pragma unroll
    for (int dt = 0; dt < 4; ++dt)
#pragma unroll
      for (int r = 0; r < 8; ++r) oacc[dt][r] *= alpha;

    // C-fragment (S' layout) -> B-fragment of P (K=32) via lane^16 exchange
    v16h pb;
    bool lo = lane < 16;
#pragma unroll
    for (int r = 0; r < 8; ++r) {
      float x0 = __shfl_xor(p0[r], 16, 32);
      float x1 = __shfl_xor(p1[r], 16, 32);
      pb[r] = (_Float16)(lo ? p0[r] : x1);
      pb[8 + r] = (_Float16)(lo ? x0 : p1[r]);
    }

    // O'[d,t] += V^T[d,s] * P'[s,t]  (V^T rows are contiguous in Vt)
#pragma unroll
    for (int dt = 0; dt < 4; ++dt)
      oacc[dt] = wmma16(frag_a(&Vt[cur][(dt * 16) * LDV], LDV), pb, oacc[dt]);

    if (more) {
#pragma unroll
      for (int j = 0; j < 8; ++j) Vt[nxt][(sc + j) * LDV + sr] = vv[j];
    }
    async_wait0();
    __syncthreads();
  }

  float inv = 1.0f / lrun;
  int rb8 = (lane >> 4) << 3;
  _Float16* op = o + (rowbase + tcol) * CC + h0;
#pragma unroll
  for (int dt = 0; dt < 4; ++dt) {
    h8 hv;
#pragma unroll
    for (int r = 0; r < 8; ++r) hv[r] = (_Float16)(oacc[dt][r] * inv);
    *(h8*)(op + dt * 16 + rb8) = hv;
  }
}

// ---------------- host orchestration --------------------------------------

extern "C" void kernel_launch(void* const* d_in, const int* in_sizes, int n_in,
                              void* d_out, int out_size, void* d_ws,
                              size_t ws_size, hipStream_t stream) {
  (void)in_sizes; (void)n_in; (void)out_size; (void)ws_size;
  const float* x      = (const float*)d_in[0];
  const float* ln1_g  = (const float*)d_in[1];
  const float* ln1_b  = (const float*)d_in[2];
  const float* wq     = (const float*)d_in[3];
  const float* wk     = (const float*)d_in[4];
  const float* wv     = (const float*)d_in[5];
  const float* w_proj = (const float*)d_in[6];
  const float* b_proj = (const float*)d_in[7];
  const float* ln2_g  = (const float*)d_in[8];
  const float* ln2_b  = (const float*)d_in[9];
  const float* w1     = (const float*)d_in[10];
  const float* b1     = (const float*)d_in[11];
  const float* w2     = (const float*)d_in[12];
  const float* b2     = (const float*)d_in[13];
  float* out = (float*)d_out;

  char* ws = (char*)d_ws;
  size_t off = 0;
  auto carve = [&](size_t bytes) {
    char* p = ws + off;
    off += (bytes + 255) & ~(size_t)255;
    return p;
  };
  const size_t actH = (size_t)BT * CC * sizeof(_Float16);  // 25,165,824
  _Float16* h16    = (_Float16*)carve(actH);  // LN1 out / LN2 out (reused)
  _Float16* q16    = (_Float16*)carve(actH);
  _Float16* k16    = (_Float16*)carve(actH);
  _Float16* v16    = (_Float16*)carve(actH);
  _Float16* attn16 = (_Float16*)carve(actH);
  float*    x1     = (float*)carve((size_t)BT * CC * sizeof(float));
  _Float16* wq16   = (_Float16*)carve((size_t)CC * CC * 2);
  _Float16* wk16   = (_Float16*)carve((size_t)CC * CC * 2);
  _Float16* wv16   = (_Float16*)carve((size_t)CC * CC * 2);
  _Float16* wp16   = (_Float16*)carve((size_t)CC * CC * 2);
  _Float16* w1_16  = (_Float16*)carve((size_t)CC * MMD * 2);
  _Float16* w2_16  = (_Float16*)carve((size_t)MMD * CC * 2);
  // mlp1 (BT x M f16 = 100,663,296 B) aliases the dead q/k/v/attn region
  _Float16* mlp1 = q16;

  const int qkvN = HH * CC * HDIM;  // 589,824

  // 1) LN1 -> f16
  layernorm_f16_k<<<BT, 256, 0, stream>>>(x, ln1_g, ln1_b, h16);

  // 2) weight conversion / packing
  pack_qkv_k<<<(qkvN + 255) / 256, 256, 0, stream>>>(wq, wq16);
  pack_qkv_k<<<(qkvN + 255) / 256, 256, 0, stream>>>(wk, wk16);
  pack_qkv_k<<<(qkvN + 255) / 256, 256, 0, stream>>>(wv, wv16);
  cvt_f16_k<<<(CC * CC + 255) / 256, 256, 0, stream>>>(w_proj, wp16, CC * CC);
  cvt_f16_k<<<(CC * MMD + 255) / 256, 256, 0, stream>>>(w1, w1_16, CC * MMD);
  cvt_f16_k<<<(MMD * CC + 255) / 256, 256, 0, stream>>>(w2, w2_16, MMD * CC);

  // 3) Q/K/V projections: [BT,768] x [768,768]
  dim3 gQ(CC / 64, BT / 128);
  gemm_wmma<0><<<gQ, 256, 0, stream>>>(h16, wq16, nullptr, nullptr, nullptr,
                                       q16, BT, CC, CC);
  gemm_wmma<0><<<gQ, 256, 0, stream>>>(h16, wk16, nullptr, nullptr, nullptr,
                                       k16, BT, CC, CC);
  gemm_wmma<0><<<gQ, 256, 0, stream>>>(h16, wv16, nullptr, nullptr, nullptr,
                                       v16, BT, CC, CC);

  // 4) flash attention -> attn16 [BT, C] (heads concatenated)
  attention_k<<<BB * HH * (TT / 128), 256, 0, stream>>>(q16, k16, v16, attn16);

  // 5) output projection + bias + residual -> x1 (f32)
  gemm_wmma<1><<<gQ, 256, 0, stream>>>(attn16, wp16, b_proj, x, x1, nullptr,
                                       BT, CC, CC);

  // 6) LN2 -> f16 (reuse h16)
  layernorm_f16_k<<<BT, 256, 0, stream>>>(x1, ln2_g, ln2_b, h16);

  // 7) MLP fc1 + exact GELU -> mlp1 f16: [BT,768] x [768,3072]
  dim3 gM1(MMD / 64, BT / 128);
  gemm_wmma<2><<<gM1, 256, 0, stream>>>(h16, w1_16, b1, nullptr, nullptr, mlp1,
                                        BT, MMD, CC);

  // 8) MLP fc2 + bias + residual -> out f32: [BT,3072] x [3072,768]
  gemm_wmma<1><<<gQ, 256, 0, stream>>>(mlp1, w2_16, b2, x1, out, nullptr, BT,
                                       CC, MMD);
}
